// DiffusionNet_57569741636140
// MI455X (gfx1250) — matrix operand order repaired and checked by reading, not compiled, not run on hardware
//
#include <hip/hip_runtime.h>
#include <cmath>

// ---------------------------------------------------------------------------
// DiffusionNet forward for gfx1250 (MI455X).
// All large GEMMs run through v_wmma_f32_16x16x32_f16 (fp32 -> f16 in LDS,
// fp32 accumulate). LN / softmax / elementwise stay fp32 VALU.
//
// GEMM kernel assumes exact tiling (all shapes in this model satisfy it):
//   M % 128 == 0, N % 64 == 0, K % 32 == 0
// ---------------------------------------------------------------------------

#define Dd   1024
#define Aa   256
#define Hh   16
#define DKk  64
#define Bbv  2
#define Llv  512
#define Natv 2048
#define BLv  (Bbv * Llv)     // 1024
#define BNv  (Bbv * Natv)    // 4096

typedef __attribute__((ext_vector_type(16))) _Float16 v16h;
typedef __attribute__((ext_vector_type(8)))  float    v8f;

union Frag16 { v16h v; unsigned int u[8]; };
union H8     { _Float16 h[8]; uint4 u4; };

#define AS_STRIDE 40   // halves per A row (32 + 8 pad)
#define BS_STRIDE 40   // halves per B row

// Registers holding one in-flight staging tile (A: 16 floats, B: 8 floats).
struct StageRegs { float4 a0, a1, a2, a3, b0, b1; };

// Phase 1: issue global loads for the 128x32 A tile / 64x32 B tile.
__device__ __forceinline__ StageRegs stage_load(
    const float* __restrict__ Ag, const float* __restrict__ Bg,
    int lda, int ldb, int mTile, int nTile, int k0, int transB, int tid)
{
    StageRegs r;
    const int arow = tid >> 1;           // 0..127
    const int akc  = (tid & 1) * 16;
    const float4* asrc =
        (const float4*)(Ag + (long long)(mTile + arow) * lda + k0 + akc);
    r.a0 = asrc[0]; r.a1 = asrc[1]; r.a2 = asrc[2]; r.a3 = asrc[3];

    if (transB) {                        // B is N x K: coalesced rows
        const int brow = tid >> 2;       // n: 0..63
        const int bkc  = (tid & 3) * 8;
        const float4* bsrc =
            (const float4*)(Bg + (long long)(nTile + brow) * ldb + k0 + bkc);
        r.b0 = bsrc[0]; r.b1 = bsrc[1];
    } else {                             // B is K x N: coalesced k-major rows
        const int bkk = tid >> 3;        // k: 0..31
        const int bnc = (tid & 7) * 8;   // n: 0..56
        const float4* bsrc =
            (const float4*)(Bg + (long long)(k0 + bkk) * ldb + nTile + bnc);
        r.b0 = bsrc[0]; r.b1 = bsrc[1];
    }
    return r;
}

// Phase 2: convert fp32 -> f16 and store into LDS ([m][k] for A, [n][k] for B).
__device__ __forceinline__ void stage_store(
    _Float16* __restrict__ As, _Float16* __restrict__ Bs,
    const StageRegs& r, int transB, int tid)
{
    {
        const int row = tid >> 1;
        const int kc  = (tid & 1) * 16;
        H8 p0, p1;
        p0.h[0] = (_Float16)r.a0.x; p0.h[1] = (_Float16)r.a0.y;
        p0.h[2] = (_Float16)r.a0.z; p0.h[3] = (_Float16)r.a0.w;
        p0.h[4] = (_Float16)r.a1.x; p0.h[5] = (_Float16)r.a1.y;
        p0.h[6] = (_Float16)r.a1.z; p0.h[7] = (_Float16)r.a1.w;
        p1.h[0] = (_Float16)r.a2.x; p1.h[1] = (_Float16)r.a2.y;
        p1.h[2] = (_Float16)r.a2.z; p1.h[3] = (_Float16)r.a2.w;
        p1.h[4] = (_Float16)r.a3.x; p1.h[5] = (_Float16)r.a3.y;
        p1.h[6] = (_Float16)r.a3.z; p1.h[7] = (_Float16)r.a3.w;
        *(uint4*)&As[row * AS_STRIDE + kc]     = p0.u4;   // ds_store_b128
        *(uint4*)&As[row * AS_STRIDE + kc + 8] = p1.u4;   // ds_store_b128
    }
    if (transB) {
        const int row = tid >> 2;
        const int kc  = (tid & 3) * 8;
        H8 p;
        p.h[0] = (_Float16)r.b0.x; p.h[1] = (_Float16)r.b0.y;
        p.h[2] = (_Float16)r.b0.z; p.h[3] = (_Float16)r.b0.w;
        p.h[4] = (_Float16)r.b1.x; p.h[5] = (_Float16)r.b1.y;
        p.h[6] = (_Float16)r.b1.z; p.h[7] = (_Float16)r.b1.w;
        *(uint4*)&Bs[row * BS_STRIDE + kc] = p.u4;        // ds_store_b128
    } else {
        const int kk = tid >> 3;
        const int nc = (tid & 7) * 8;
        Bs[(nc + 0) * BS_STRIDE + kk] = (_Float16)r.b0.x;
        Bs[(nc + 1) * BS_STRIDE + kk] = (_Float16)r.b0.y;
        Bs[(nc + 2) * BS_STRIDE + kk] = (_Float16)r.b0.z;
        Bs[(nc + 3) * BS_STRIDE + kk] = (_Float16)r.b0.w;
        Bs[(nc + 4) * BS_STRIDE + kk] = (_Float16)r.b1.x;
        Bs[(nc + 5) * BS_STRIDE + kk] = (_Float16)r.b1.y;
        Bs[(nc + 6) * BS_STRIDE + kk] = (_Float16)r.b1.z;
        Bs[(nc + 7) * BS_STRIDE + kk] = (_Float16)r.b1.w;
    }
}

// ---------------------------------------------------------------------------
// Batched WMMA GEMM:  C[b] = alpha * A[b] x op(B[b]) + bias
//   Block: 256 thr = 8 waves; tile 128(M) x 64(N); K-step 32; LDS double-buffer
//   with software-pipelined staging (global loads issued before the WMMAs,
//   cvt+LDS-store after, so global latency hides behind the matrix pipe).
//   Wave grid 4x2; each wave computes 32x32 = 4 x v_wmma_f32_16x16x32_f16.
// ---------------------------------------------------------------------------
__global__ __launch_bounds__(256) void gemm_wmma(
    const float* __restrict__ Ag, int lda, long long sAi, long long sAo,
    const float* __restrict__ Bg, int ldb, long long sBi, long long sBo,
    float* __restrict__ Cg, int ldc, long long sCi, long long sCo,
    int M, int N, int K, int batch_in, int transB, float alpha,
    const float* __restrict__ bias)
{
    __shared__ _Float16 As[2][128 * AS_STRIDE];   // 2 x 10 KB
    __shared__ _Float16 Bs[2][64 * BS_STRIDE];    // 2 x 5 KB

    const int z  = blockIdx.z;
    const int zi = z % batch_in;
    const int zo = z / batch_in;
    Ag += (long long)zi * sAi + (long long)zo * sAo;
    Bg += (long long)zi * sBi + (long long)zo * sBo;
    Cg += (long long)zi * sCi + (long long)zo * sCo;

    const int mTile = blockIdx.y * 128;
    const int nTile = blockIdx.x * 64;

    const int tid  = threadIdx.x;
    const int lane = tid & 31;
    const int wave = tid >> 5;
    const int wm   = (wave & 3) * 32;   // wave M offset in tile
    const int wn   = (wave >> 2) * 32;  // wave N offset in tile

    v8f acc00 = {}, acc01 = {}, acc10 = {}, acc11 = {};

    const int nsteps = K >> 5;

    // prologue: stage tile 0
    {
        StageRegs r0 = stage_load(Ag, Bg, lda, ldb, mTile, nTile, 0, transB, tid);
        stage_store(As[0], Bs[0], r0, transB, tid);
    }
    __syncthreads();

    for (int s = 0; s < nsteps; ++s) {
        const int cur = s & 1;
        const int nxt = cur ^ 1;
        const int k0  = s << 5;

        // issue next tile's global loads BEFORE the math (latency hiding)
        StageRegs regs;
        if (s + 1 < nsteps)
            regs = stage_load(Ag, Bg, lda, ldb, mTile, nTile, k0 + 32, transB, tid);
        if (s + 2 < nsteps)   // emits global_prefetch_b8
            __builtin_prefetch(Ag + (long long)(mTile + (tid >> 1)) * lda + k0 + 64, 0, 3);

        // ---- A fragments (16x32 f16, ISA wave32 layout) ----
        Frag16 fa0, fa1;
        const int ar0 = wm + (lane & 15);
        const int ar1 = ar0 + 16;
        const int kb8 = (lane >> 4) * 8;   // lanes 16-31 shifted by 8 in K
#pragma unroll
        for (int r = 0; r < 8; ++r) {
            const int kk = ((r & 4) ? 16 : 0) + kb8 + 2 * (r & 3);
            fa0.u[r] = *(const unsigned int*)&As[cur][ar0 * AS_STRIDE + kk];
            fa1.u[r] = *(const unsigned int*)&As[cur][ar1 * AS_STRIDE + kk];
        }
        // ---- B fragments (32x16 f16: lane = N, K in same 8-interleaved order as A) ----
        Frag16 fb0, fb1;
        const int bn0  = wn + (lane & 15);
        const int bn1  = bn0 + 16;
        const int kb16 = (lane >> 4) * 16;
#pragma unroll
        for (int r = 0; r < 8; ++r) {
            fb0.u[r] = *(const unsigned int*)&Bs[cur][bn0 * BS_STRIDE + ((r & 4) ? 16 : 0) + kb8 + 2 * (r & 3)];
            fb1.u[r] = *(const unsigned int*)&Bs[cur][bn1 * BS_STRIDE + ((r & 4) ? 16 : 0) + kb8 + 2 * (r & 3)];
        }

        acc00 = __builtin_amdgcn_wmma_f32_16x16x32_f16(false, fa0.v, false, fb0.v,
                                                       (short)0, acc00, false, false);
        acc01 = __builtin_amdgcn_wmma_f32_16x16x32_f16(false, fa0.v, false, fb1.v,
                                                       (short)0, acc01, false, false);
        acc10 = __builtin_amdgcn_wmma_f32_16x16x32_f16(false, fa1.v, false, fb0.v,
                                                       (short)0, acc10, false, false);
        acc11 = __builtin_amdgcn_wmma_f32_16x16x32_f16(false, fa1.v, false, fb1.v,
                                                       (short)0, acc11, false, false);

        // convert + store next tile AFTER the math
        if (s + 1 < nsteps)
            stage_store(As[nxt], Bs[nxt], regs, transB, tid);
        __syncthreads();
    }

    // ---- epilogue: C 16x16 f32 layout (lanes 16-31 hold M=8..15) ----
    const int em0 = mTile + wm + ((lane >> 4) * 8);
    const int em1 = em0 + 16;
    const int en0 = nTile + wn + (lane & 15);
    const int en1 = en0 + 16;
    const float bv0 = bias ? bias[en0] : 0.f;
    const float bv1 = bias ? bias[en1] : 0.f;
#pragma unroll
    for (int r = 0; r < 8; ++r) {
        Cg[(long long)(em0 + r) * ldc + en0] = acc00[r] * alpha + bv0;
        Cg[(long long)(em0 + r) * ldc + en1] = acc01[r] * alpha + bv1;
        Cg[(long long)(em1 + r) * ldc + en0] = acc10[r] * alpha + bv0;
        Cg[(long long)(em1 + r) * ldc + en1] = acc11[r] * alpha + bv1;
    }
}

// ---------------------------------------------------------------------------
// Scalar GEMM for tiny K or N (to_coords, to_confs, coord_embed, out_w)
// ---------------------------------------------------------------------------
__global__ void gemm_small(const float* __restrict__ A, int lda,
                           const float* __restrict__ W, int ldw,
                           float* __restrict__ C, int ldc,
                           int M, int N, int K, float alpha,
                           const float* __restrict__ bias)
{
    long long total = (long long)M * N;
    for (long long idx = (long long)blockIdx.x * blockDim.x + threadIdx.x;
         idx < total; idx += (long long)gridDim.x * blockDim.x) {
        const int m = (int)(idx / N);
        const int n = (int)(idx % N);
        float s = 0.f;
        for (int k = 0; k < K; ++k)
            s += A[(long long)m * lda + k] * W[(long long)k * ldw + n];
        s *= alpha;
        if (bias) s += bias[n];
        C[(long long)m * ldc + n] = s;
    }
}

// ---------------------------------------------------------------------------
// Row LayerNorm (block=256, one block per row)
// ---------------------------------------------------------------------------
__global__ __launch_bounds__(256) void ln_kernel(
    const float* __restrict__ x, const float* __restrict__ g,
    const float* __restrict__ b, float* __restrict__ y, int dim)
{
    const int row = blockIdx.x;
    const float* xr = x + (long long)row * dim;
    float* yr = y + (long long)row * dim;
    __shared__ float red[256];
    const int tid = threadIdx.x;

    float s = 0.f;
    for (int i = tid; i < dim; i += 256) s += xr[i];
    red[tid] = s; __syncthreads();
    for (int off = 128; off > 0; off >>= 1) {
        if (tid < off) red[tid] += red[tid + off];
        __syncthreads();
    }
    const float mean = red[0] / dim;
    __syncthreads();

    float v = 0.f;
    for (int i = tid; i < dim; i += 256) { float d = xr[i] - mean; v += d * d; }
    red[tid] = v; __syncthreads();
    for (int off = 128; off > 0; off >>= 1) {
        if (tid < off) red[tid] += red[tid + off];
        __syncthreads();
    }
    const float inv = rsqrtf(red[0] / dim + 1e-5f);

    for (int i = tid; i < dim; i += 256)
        yr[i] = (xr[i] - mean) * inv * g[i] + b[i];
}

// ---------------------------------------------------------------------------
// Row softmax, optional ALiBi bias (slope = 2^(-0.5*(h+1)) for 16 heads)
// row index = ((b*nheads + h)*rowlen + i) when alibi != 0
// ---------------------------------------------------------------------------
__global__ __launch_bounds__(256) void softmax_kernel(
    float* __restrict__ s, int rowlen, int nheads, int alibi)
{
    const long long row = blockIdx.x;
    float* sr = s + row * rowlen;
    float slope = 0.f;
    int i0 = 0;
    if (alibi) {
        i0 = (int)(row % rowlen);
        const int hh = (int)((row / rowlen) % nheads);
        slope = exp2f(-0.5f * (float)(hh + 1));
    }
    __shared__ float red[256];
    const int tid = threadIdx.x;

    float m = -3.4e38f;
    for (int j = tid; j < rowlen; j += 256) {
        float v = sr[j];
        if (alibi) v -= slope * fabsf((float)(i0 - j));
        m = fmaxf(m, v);
    }
    red[tid] = m; __syncthreads();
    for (int off = 128; off > 0; off >>= 1) {
        if (tid < off) red[tid] = fmaxf(red[tid], red[tid + off]);
        __syncthreads();
    }
    const float rowmax = red[0];
    __syncthreads();

    float sum = 0.f;
    for (int j = tid; j < rowlen; j += 256) {
        float v = sr[j];
        if (alibi) v -= slope * fabsf((float)(i0 - j));
        const float e = __expf(v - rowmax);
        sr[j] = e;
        sum += e;
    }
    red[tid] = sum; __syncthreads();
    for (int off = 128; off > 0; off >>= 1) {
        if (tid < off) red[tid] += red[tid + off];
        __syncthreads();
    }
    const float inv = 1.f / red[0];
    for (int j = tid; j < rowlen; j += 256) sr[j] *= inv;
}

// ---------------------------------------------------------------------------
// Elementwise kernels
// ---------------------------------------------------------------------------
__global__ void gate_residual_kernel(float* __restrict__ x,
                                     const float* __restrict__ gate,
                                     const float* __restrict__ o, long long n)
{
    for (long long i = (long long)blockIdx.x * blockDim.x + threadIdx.x;
         i < n; i += (long long)gridDim.x * blockDim.x) {
        const float sg = 1.f / (1.f + __expf(-gate[i]));
        x[i] = x[i] + sg * o[i];
    }
}

__global__ void add_kernel(float* __restrict__ x, const float* __restrict__ y, long long n)
{
    for (long long i = (long long)blockIdx.x * blockDim.x + threadIdx.x;
         i < n; i += (long long)gridDim.x * blockDim.x)
        x[i] += y[i];
}

__global__ void swiglu_kernel(const float* __restrict__ in, float* __restrict__ out,
                              int rows, int H)
{
    long long total = (long long)rows * H;
    for (long long idx = (long long)blockIdx.x * blockDim.x + threadIdx.x;
         idx < total; idx += (long long)gridDim.x * blockDim.x) {
        const int r = (int)(idx / H);
        const int j = (int)(idx % H);
        const float a  = in[(long long)r * 2 * H + j];
        const float gt = in[(long long)r * 2 * H + H + j];
        out[idx] = a * gt / (1.f + __expf(-gt));
    }
}

// f = [sin(t/2^i), cos(t/2^i)] (i=0..7) ; out[b][c] = f . w[:,c] + bias[c]
__global__ __launch_bounds__(256) void fourier_kernel(
    const float* __restrict__ nlev, const float* __restrict__ w,
    const float* __restrict__ bb, float* __restrict__ out)
{
    const int b = blockIdx.x;
    const int c = threadIdx.x;       // A = 256
    const float t = nlev[b];
    float acc = bb[c];
#pragma unroll
    for (int i = 0; i < 8; ++i) {
        const float xs = t / exp2f((float)i);
        acc += sinf(xs) * w[i * Aa + c] + cosf(xs) * w[(i + 8) * Aa + c];
    }
    out[b * Aa + c] = acc;
}

__global__ void scale_coords_kernel(const float* __restrict__ noised,
                                    const float* __restrict__ nlev,
                                    float* __restrict__ out, int Bn, int Nn)
{
    long long total = (long long)Bn * Nn * 3;
    for (long long idx = (long long)blockIdx.x * blockDim.x + threadIdx.x;
         idx < total; idx += (long long)gridDim.x * blockDim.x) {
        const int b = (int)(idx / ((long long)Nn * 3));
        const float t = nlev[b];
        out[idx] = noised[idx] * rsqrtf(t * t + 256.0f);
    }
}

// atom_x = a[b, aaidx[n]] + aa_embed[aacodes[aaidx[n]]] + atom_embed[atcodes[n]]
//        + pos_encoding(aaidx[n], A) + fourier[b] + coord_proj[b,n]
__global__ void assemble_atom_kernel(
    const float* __restrict__ abuf, const int* __restrict__ aaidx,
    const int* __restrict__ aacodes, const int* __restrict__ atcodes,
    const float* __restrict__ aa_embed, const float* __restrict__ at_embed,
    const float* __restrict__ fe, const float* __restrict__ cproj,
    float* __restrict__ outx, int Bn, int Nn, int Ll)
{
    long long total = (long long)Bn * Nn * Aa;
    for (long long idx = (long long)blockIdx.x * blockDim.x + threadIdx.x;
         idx < total; idx += (long long)gridDim.x * blockDim.x) {
        const int c = (int)(idx % Aa);
        const int n = (int)((idx / Aa) % Nn);
        const int b = (int)(idx / ((long long)Aa * Nn));
        const int ai = aaidx[n];
        const int j  = c >> 1;
        const float div = __expf(-logf(10000.f) * (2.f * (float)j) / (float)Aa);
        const float ang = (float)ai * div;
        const float pe  = (c & 1) ? cosf(ang) : sinf(ang);
        outx[idx] = abuf[((long long)b * Ll + ai) * Aa + c]
                  + aa_embed[aacodes[ai] * Aa + c]
                  + at_embed[atcodes[n] * Aa + c]
                  + pe
                  + fe[b * Aa + c]
                  + cproj[((long long)b * Nn + n) * Aa + c];
    }
}

// pred_denoised = den * SIGDATA*t/sqrt(VAR+t^2) + noised * VAR/(VAR+t^2)
__global__ void denoise_kernel(const float* __restrict__ den,
                               const float* __restrict__ noised,
                               const float* __restrict__ nlev,
                               float* __restrict__ out, int Bn, int Nn)
{
    long long total = (long long)Bn * Nn * 3;
    for (long long idx = (long long)blockIdx.x * blockDim.x + threadIdx.x;
         idx < total; idx += (long long)gridDim.x * blockDim.x) {
        const int b = (int)(idx / ((long long)Nn * 3));
        const float t  = nlev[b];
        const float d2 = 256.0f + t * t;
        out[idx] = den[idx] * (16.0f * t / sqrtf(d2)) + noised[idx] * (256.0f / d2);
    }
}

// ---------------------------------------------------------------------------
// Host orchestration
// ---------------------------------------------------------------------------
extern "C" void kernel_launch(void* const* d_in, const int* in_sizes, int n_in,
                              void* d_out, int out_size, void* d_ws, size_t ws_size,
                              hipStream_t stream)
{
    (void)in_sizes; (void)out_size; (void)ws_size;
    if (n_in < 251) return;   // expect 6 inputs + 245 param leaves

    auto F = [&](int i) { return (const float*)d_in[i]; };
    auto I = [&](int i) { return (const int*)d_in[i]; };

    const float* x_in    = F(0);
    const int*   aacodes = I(1);
    const int*   atcodes = I(2);
    const int*   aaidx   = I(3);
    const float* noised  = F(4);
    const float* nlev    = F(5);

    int pi = 6;
    const float* norm1_g = F(pi++); const float* norm1_b = F(pi++);
    struct SP { const float *wq,*wk,*wv,*wo,*gw,*gb,*ff1,*ff2,*ln1g,*ln1b,*ln2g,*ln2b; } sp[8];
    for (int i = 0; i < 8; ++i) {
        sp[i].wq = F(pi++); sp[i].wk = F(pi++); sp[i].wv = F(pi++); sp[i].wo = F(pi++);
        sp[i].gw = F(pi++); sp[i].gb = F(pi++); sp[i].ff1 = F(pi++); sp[i].ff2 = F(pi++);
        sp[i].ln1g = F(pi++); sp[i].ln1b = F(pi++); sp[i].ln2g = F(pi++); sp[i].ln2b = F(pi++);
    }
    const float* to_coords = F(pi++);
    const float* to_confs  = F(pi++);
    const float* to_atom   = F(pi++);
    const float* norm2_g = F(pi++); const float* norm2_b = F(pi++);
    const float* norm3_g = F(pi++); const float* norm3_b = F(pi++);
    const float* aa_embed   = F(pi++);
    const float* atom_embed = F(pi++);
    const float* coord_embed = F(pi++);
    const float* nw = F(pi++); const float* nb = F(pi++);
    const float* ng = F(pi++); const float* nbeta = F(pi++);
    struct AP { const float *wq,*wk,*wv,*gw,*gb,*ff1,*ff1b,*ff2,*ff2b,*ln1g,*ln1b,*ln2g,*ln2b; } ap[10];
    for (int i = 0; i < 10; ++i) {
        ap[i].wq = F(pi++); ap[i].wk = F(pi++); ap[i].wv = F(pi++);
        ap[i].gw = F(pi++); ap[i].gb = F(pi++);
        ap[i].ff1 = F(pi++); ap[i].ff1b = F(pi++);
        ap[i].ff2 = F(pi++); ap[i].ff2b = F(pi++);
        ap[i].ln1g = F(pi++); ap[i].ln1b = F(pi++); ap[i].ln2g = F(pi++); ap[i].ln2b = F(pi++);
    }
    const float* out_ln_g = F(pi++); const float* out_ln_b = F(pi++);
    const float* out_w    = F(pi++);

    // ---- workspace bump allocator (~135 MB needed) ----
    char* ws = (char*)d_ws;
    size_t off = 0;
    auto alloc = [&](size_t nfl) {
        float* p = (float*)(ws + off);
        off += nfl * sizeof(float);
        off = (off + 255) & ~(size_t)255;
        return p;
    };
    float* xb    = alloc((size_t)BLv * Dd);
    float* hn    = alloc((size_t)BLv * Dd);
    float* qb    = alloc((size_t)BLv * Dd);
    float* kb    = alloc((size_t)BLv * Dd);
    float* vb    = alloc((size_t)BLv * Dd);
    float* gate  = alloc((size_t)BLv * Dd);
    float* sc    = alloc((size_t)Bbv * Hh * Llv * Llv);   // also holds atom scores
    float* oh    = alloc((size_t)BLv * Dd);
    float* ob    = alloc((size_t)BLv * Dd);
    float* ff1b  = alloc((size_t)BLv * 8 * Dd);
    float* swb   = alloc((size_t)BLv * 4 * Dd);
    float* ff2o  = alloc((size_t)BLv * Dd);
    float* abuf  = alloc((size_t)BLv * Aa);
    float* ax    = alloc((size_t)BNv * Aa);
    float* axw   = alloc((size_t)BNv * Aa);
    float* fe    = alloc((size_t)Bbv * Aa);
    float* cproj = alloc((size_t)BNv * Aa);
    float* csc   = alloc((size_t)BNv * 3);
    float* den   = alloc((size_t)BNv * 3);

    float* outp = (float*)d_out;
    float* out_den    = outp;                 // [B,N,3]  12288
    float* out_coords = outp + 12288;         // [B,L,3]   3072
    float* out_confs  = outp + 15360;         // [B,L]     1024

    auto GEMM = [&](const float* Ap, int lda, long long sAi, long long sAo,
                    const float* Bp, int ldb, long long sBi, long long sBo,
                    float* Cp, int ldc, long long sCi, long long sCo,
                    int M, int N, int K, int bi, int bo, int transB,
                    float alpha, const float* bias) {
        dim3 g((N + 63) / 64, (M + 127) / 128, (unsigned)(bi * bo));
        gemm_wmma<<<g, dim3(256), 0, stream>>>(Ap, lda, sAi, sAo, Bp, ldb, sBi, sBo,
                                               Cp, ldc, sCi, sCo, M, N, K, bi, transB,
                                               alpha, bias);
    };
    auto LN = [&](const float* xp, const float* gp, const float* bp, float* yp,
                  int rows, int dim) {
        ln_kernel<<<dim3((unsigned)rows), dim3(256), 0, stream>>>(xp, gp, bp, yp, dim);
    };
    auto SOFTMAX = [&](float* s, long long rows, int rowlen, int nheads, int alibi) {
        softmax_kernel<<<dim3((unsigned)rows), dim3(256), 0, stream>>>(s, rowlen, nheads, alibi);
    };
    auto EW = [&](long long n) { return dim3((unsigned)((n + 255) / 256)); };

    const float sSeq  = powf((float)DKk, -0.25f);        // (D/h)^-0.25
    const float sAtom = powf((float)(Aa / 8), -0.25f);   // 32^-0.25

    // ============================ sequence trunk ============================
    LN(x_in, norm1_g, norm1_b, xb, BLv, Dd);

    for (int cyc = 0; cyc < 3; ++cyc) {
        for (int l = 0; l < 8; ++l) {
            const SP& p = sp[l];
            LN(xb, p.ln1g, p.ln1b, hn, BLv, Dd);
            GEMM(hn, Dd, 0, 0, p.wq, Dd, 0, 0, qb, Dd, 0, 0, BLv, Dd, Dd, 1, 1, 0, sSeq, nullptr);
            GEMM(hn, Dd, 0, 0, p.wk, Dd, 0, 0, kb, Dd, 0, 0, BLv, Dd, Dd, 1, 1, 0, sSeq, nullptr);
            GEMM(hn, Dd, 0, 0, p.wv, Dd, 0, 0, vb, Dd, 0, 0, BLv, Dd, Dd, 1, 1, 0, 1.f, nullptr);
            GEMM(hn, Dd, 0, 0, p.gw, Dd, 0, 0, gate, Dd, 0, 0, BLv, Dd, Dd, 1, 1, 0, 1.f, p.gb);
            // scores[b,h] = q[b,:,h,:] x k[b,:,h,:]^T
            GEMM(qb, Dd, DKk, (long long)Llv * Dd,
                 kb, Dd, DKk, (long long)Llv * Dd,
                 sc, Llv, (long long)Llv * Llv, (long long)Hh * Llv * Llv,
                 Llv, Llv, DKk, Hh, Bbv, 1, 1.f, nullptr);
            SOFTMAX(sc, (long long)Bbv * Hh * Llv, Llv, Hh, 1);
            // o_heads[b,h] = attn x v[b,:,h,:]
            GEMM(sc, Llv, (long long)Llv * Llv, (long long)Hh * Llv * Llv,
                 vb, Dd, DKk, (long long)Llv * Dd,
                 oh, Dd, DKk, (long long)Llv * Dd,
                 Llv, DKk, Llv, Hh, Bbv, 0, 1.f, nullptr);
            GEMM(oh, Dd, 0, 0, p.wo, Dd, 0, 0, ob, Dd, 0, 0, BLv, Dd, Dd, 1, 1, 0, 1.f, nullptr);
            gate_residual_kernel<<<EW((long long)BLv * Dd), 256, 0, stream>>>(xb, gate, ob, (long long)BLv * Dd);

            LN(xb, p.ln2g, p.ln2b, hn, BLv, Dd);
            GEMM(hn, Dd, 0, 0, p.ff1, 8 * Dd, 0, 0, ff1b, 8 * Dd, 0, 0, BLv, 8 * Dd, Dd, 1, 1, 0, 1.f, nullptr);
            swiglu_kernel<<<EW((long long)BLv * 4 * Dd), 256, 0, stream>>>(ff1b, swb, BLv, 4 * Dd);
            GEMM(swb, 4 * Dd, 0, 0, p.ff2, Dd, 0, 0, ff2o, Dd, 0, 0, BLv, Dd, 4 * Dd, 1, 1, 0, 1.f, nullptr);
            add_kernel<<<EW((long long)BLv * Dd), 256, 0, stream>>>(xb, ff2o, (long long)BLv * Dd);
        }
    }

    // ============================ heads + atom embed ========================
    gemm_small<<<EW((long long)BLv * 3), 256, 0, stream>>>(xb, Dd, to_coords, 3, out_coords, 3, BLv, 3, Dd, 1.f, nullptr);
    gemm_small<<<EW((long long)BLv), 256, 0, stream>>>(xb, Dd, to_confs, 1, out_confs, 1, BLv, 1, Dd, 1.f, nullptr);

    GEMM(xb, Dd, 0, 0, to_atom, Aa, 0, 0, qb, Aa, 0, 0, BLv, Aa, Dd, 1, 1, 0, 1.f, nullptr);
    LN(qb, norm2_g, norm2_b, abuf, BLv, Aa);

    fourier_kernel<<<dim3(Bbv), dim3(Aa), 0, stream>>>(nlev, nw, nb, kb);
    LN(kb, ng, nbeta, fe, Bbv, Aa);

    scale_coords_kernel<<<EW((long long)BNv * 3), 256, 0, stream>>>(noised, nlev, csc, Bbv, Natv);
    gemm_small<<<EW((long long)BNv * Aa), 256, 0, stream>>>(csc, 3, coord_embed, Aa, cproj, Aa, BNv, Aa, 3, 1.f, nullptr);

    assemble_atom_kernel<<<EW((long long)BNv * Aa), 256, 0, stream>>>(
        abuf, aaidx, aacodes, atcodes, aa_embed, atom_embed, fe, cproj, ax, Bbv, Natv, Llv);
    LN(ax, norm3_g, norm3_b, axw, BNv, Aa);

    // ============================ atom trunk ================================
    for (int l = 0; l < 10; ++l) {
        const AP& p = ap[l];
        LN(axw, p.ln1g, p.ln1b, hn, BNv, Aa);
        GEMM(hn, Aa, 0, 0, p.wq, Aa, 0, 0, qb, Aa, 0, 0, BNv, Aa, Aa, 1, 1, 0, sAtom, nullptr);
        GEMM(hn, Aa, 0, 0, p.wk, Aa, 0, 0, kb, Aa, 0, 0, BNv, Aa, Aa, 1, 1, 0, sAtom, nullptr);
        GEMM(hn, Aa, 0, 0, p.wv, Aa, 0, 0, vb, Aa, 0, 0, BNv, Aa, Aa, 1, 1, 0, 1.f, nullptr);
        GEMM(hn, Aa, 0, 0, p.gw, Aa, 0, 0, gate, Aa, 0, 0, BNv, Aa, Aa, 1, 1, 0, 1.f, p.gb);
        // scores[b] = q[b] x k[b]^T  (2048 x 2048, K=256)
        GEMM(qb, Aa, 0, (long long)Natv * Aa,
             kb, Aa, 0, (long long)Natv * Aa,
             sc, Natv, 0, (long long)Natv * Natv,
             Natv, Natv, Aa, 1, Bbv, 1, 1.f, nullptr);
        SOFTMAX(sc, (long long)Bbv * Natv, Natv, 1, 0);
        GEMM(sc, Natv, 0, (long long)Natv * Natv,
             vb, Aa, 0, (long long)Natv * Aa,
             oh, Aa, 0, (long long)Natv * Aa,
             Natv, Aa, Natv, 1, Bbv, 0, 1.f, nullptr);
        gate_residual_kernel<<<EW((long long)BNv * Aa), 256, 0, stream>>>(axw, gate, oh, (long long)BNv * Aa);

        LN(axw, p.ln2g, p.ln2b, hn, BNv, Aa);
        GEMM(hn, Aa, 0, 0, p.ff1, 4 * Aa, 0, 0, ff1b, 4 * Aa, 0, 0, BNv, 4 * Aa, Aa, 1, 1, 0, 1.f, p.ff1b);
        swiglu_kernel<<<EW((long long)BNv * 2 * Aa), 256, 0, stream>>>(ff1b, swb, BNv, 2 * Aa);
        GEMM(swb, 2 * Aa, 0, 0, p.ff2, Aa, 0, 0, ff2o, Aa, 0, 0, BNv, Aa, 2 * Aa, 1, 1, 0, 1.f, p.ff2b);
        add_kernel<<<EW((long long)BNv * Aa), 256, 0, stream>>>(axw, ff2o, (long long)BNv * Aa);
    }

    // ============================ denoising head ============================
    LN(axw, out_ln_g, out_ln_b, hn, BNv, Aa);
    gemm_small<<<EW((long long)BNv * 3), 256, 0, stream>>>(hn, Aa, out_w, 3, den, 3, BNv, 3, Aa, 1.f, nullptr);
    denoise_kernel<<<EW((long long)BNv * 3), 256, 0, stream>>>(den, noised, nlev, out_den, Bbv, Natv);
}
